// AttLoc_48765058679363
// MI455X (gfx1250) — compile-verified, matrix operand order checked by reading, not run on hardware
//
#include <hip/hip_runtime.h>
#include <hip/hip_bf16.h>

// ---------------------------------------------------------------------------
// Problem constants (match reference)
#define B_   32
#define T_   2048
#define E_   1024   // EPROJS  (GEMM K main part)
#define D_   1024   // ATT_DIM (GEMM N)
#define NCH  10     // ACONV_CHANS
#define FLT  201    // 2*ACONV_FILTS+1
#define PADF 100
#define SCALING 2.0f

#define KT_   32     // K tiles over E_ (each 32 wide)
#define NT_   64     // N tiles (each 16 wide)
#define MTILE 32     // rows (t values) per block: 2 x 16-row WMMA subtiles
#define APITCH 1064  // ushorts per LDS A row (1056 + 8 pad -> bank-friendly)

typedef __attribute__((ext_vector_type(16))) __bf16 v16bf;
typedef __attribute__((ext_vector_type(8)))  float  v8f;

union ABfrag {
    uint4 q[2];
    unsigned short s[16];
    v16bf v;
};

__device__ __forceinline__ unsigned short f2bf(float f) {
    unsigned int u = __float_as_uint(f);
    unsigned int r = u + 0x7FFFu + ((u >> 16) & 1u);   // round-to-nearest-even
    return (unsigned short)(r >> 16);
}

// ---------------------------------------------------------------------------
// 1) dec_proj[b,d] = dec_z[b,:] @ W_dec[:,d]
__global__ __launch_bounds__(256) void k_decproj(
    const float* __restrict__ dz, const float* __restrict__ Wd,
    float* __restrict__ decp)
{
    int b = blockIdx.y;
    int d = blockIdx.x * 256 + threadIdx.x;
    const float* z = dz + (size_t)b * E_;
    float s = 0.f;
    #pragma unroll 4
    for (int k = 0; k < E_; ++k) s += z[k] * Wd[(size_t)k * D_ + d];
    decp[(size_t)b * D_ + d] = s;
}

// ---------------------------------------------------------------------------
// 2) att_conv[b,t,c] = sum_k conv_w[c,k] * att_prev[b, t+k-100]  (same pad)
//    stored as acv[b][t][16] (c fastest, padded to 16)
__global__ __launch_bounds__(256) void k_conv(
    const float* __restrict__ ap, const float* __restrict__ cw,
    float* __restrict__ acv)
{
    __shared__ float sx[256 + 2 * PADF];   // 456
    __shared__ float scw[NCH * FLT];       // 2010
    int b = blockIdx.y, t0 = blockIdx.x * 256, tid = threadIdx.x;
    for (int i = tid; i < NCH * FLT; i += 256) scw[i] = cw[i];
    for (int i = tid; i < 256 + 2 * PADF; i += 256) {
        int t = t0 + i - PADF;
        sx[i] = (t >= 0 && t < T_) ? ap[(size_t)b * T_ + t] : 0.f;
    }
    __syncthreads();
    int t = t0 + tid;
    float* out = &acv[((size_t)b * T_ + t) * 16];
    #pragma unroll
    for (int c = 0; c < NCH; ++c) {
        const float* w = &scw[c * FLT];
        float s = 0.f;
        for (int k = 0; k < FLT; ++k) s += w[k] * sx[tid + k];
        out[c] = s;
    }
}

// ---------------------------------------------------------------------------
// 3) pack W_enc (f32, [k][n]) into bf16 WMMA-B fragments:
//    offset = ((nt*KT_ + kt)*32 + lane) * 32 bytes, 16 bf16 per lane
__global__ __launch_bounds__(256) void k_packW(
    const float* __restrict__ W, unsigned char* __restrict__ pW)
{
    int gid  = blockIdx.x * 256 + threadIdx.x;   // 65536 = 64nt * 32kt * 32lane
    int lane = gid & 31;
    int kt   = (gid >> 5) & 31;
    int nt   = gid >> 10;
    int l15 = lane & 15, lh = lane >> 4;
    int n  = nt * 16 + l15;
    int kb = kt * 32 + lh * 8;
    ABfrag u;
    #pragma unroll
    for (int j = 0; j < 8; ++j) u.s[j]     = f2bf(W[(size_t)(kb + j) * D_ + n]);
    #pragma unroll
    for (int j = 0; j < 8; ++j) u.s[j + 8] = f2bf(W[(size_t)(kb + 16 + j) * D_ + n]);
    uint4* dst = (uint4*)(pW + ((size_t)gid << 5));
    dst[0] = u.q[0];
    dst[1] = u.q[1];
}

// ---------------------------------------------------------------------------
// 4) e[b,t] = gvec . tanh( enc@W_enc + acv@W_att + dec_proj + b_enc )
//    WMMA bf16 GEMM with extended K tile fusing loc/dec_proj/bias.
//    2 M-subtiles per wave share each B fragment (halves L2 traffic/FLOP).
__device__ __forceinline__ float extB(int kp, const float* __restrict__ Wa,
                                      const float* __restrict__ decp_b,
                                      const float* __restrict__ benc, int n)
{
    if (kp < NCH)  return Wa[kp * D_ + n];
    if (kp == NCH) return decp_b[n] + benc[n];
    return 0.f;
}

__device__ __forceinline__ void loadA(ABfrag& a, const unsigned short* arow, int kt)
{
    a.q[0] = *(const uint4*)(arow + kt * 32);
    a.q[1] = *(const uint4*)(arow + kt * 32 + 16);
}

__device__ __forceinline__ void loadB(ABfrag& bb, const unsigned char* pW,
                                      int nt_g, int kt, int lane)
{
    const uint4* bp =
        (const uint4*)(pW + (((size_t)(nt_g * KT_ + kt) * 32 + lane) << 5));
    bb.q[0] = bp[0];
    bb.q[1] = bp[1];
}

__global__ __launch_bounds__(256) void k_escore(
    const float* __restrict__ enc,   // B,T,E
    const float* __restrict__ gvec,  // D
    const float* __restrict__ Watt,  // NCH x D
    const float* __restrict__ benc,  // D
    const float* __restrict__ decp,  // B x D   (ws)
    const float* __restrict__ acv,   // B x T x 16 (ws)
    const unsigned char* __restrict__ pW, // packed bf16 W_enc
    float* __restrict__ e_out)       // B x T
{
    __shared__ __align__(16) unsigned short lsA[MTILE * APITCH]; // ~68 KB
    __shared__ float lsE[MTILE];

    int b   = blockIdx.y;
    int t0  = blockIdx.x * MTILE;
    int tid = threadIdx.x;

    // --- stage enc tile (MTILE x 1024 f32 -> bf16 LDS) ---
    const float* encBase = enc + ((size_t)b * T_ + t0) * E_;
    for (int i = tid; i < MTILE * (E_ / 4); i += 256) {
        int row  = i / (E_ / 4);
        int col4 = i % (E_ / 4);
        float4 f = ((const float4*)(encBase + (size_t)row * E_))[col4];
        unsigned short* d = &lsA[row * APITCH + col4 * 4];
        d[0] = f2bf(f.x); d[1] = f2bf(f.y); d[2] = f2bf(f.z); d[3] = f2bf(f.w);
    }
    // --- stage extended K columns: acv (c<10), 1.0 at c==10, else 0 ---
    for (int i = tid; i < MTILE * 32; i += 256) {
        int row = i >> 5, kc = i & 31;
        float v = (kc < NCH) ? acv[((size_t)b * T_ + t0 + row) * 16 + kc]
                             : (kc == NCH ? 1.0f : 0.0f);
        lsA[row * APITCH + E_ + kc] = f2bf(v);
    }
    if (tid < MTILE) lsE[tid] = 0.f;
    __syncthreads();

    int wave = tid >> 5;           // 0..7 : owns 128 N columns
    int lane = tid & 31;
    int l15  = lane & 15;
    int lh   = lane >> 4;
    const unsigned short* arow0 = &lsA[l15 * APITCH + lh * 8];          // rows 0..15
    const unsigned short* arow1 = &lsA[(16 + l15) * APITCH + lh * 8];   // rows 16..31
    const float* decp_b = decp + (size_t)b * D_;

    float epart0[8], epart1[8];
    #pragma unroll
    for (int r = 0; r < 8; ++r) { epart0[r] = 0.f; epart1[r] = 0.f; }

    #pragma clang loop unroll(disable)
    for (int ntl = 0; ntl < 8; ++ntl) {
        int nt_g = wave * 8 + ntl;
        int n    = nt_g * 16 + l15;
        v8f acc0 = {0.f, 0.f, 0.f, 0.f, 0.f, 0.f, 0.f, 0.f};
        v8f acc1 = {0.f, 0.f, 0.f, 0.f, 0.f, 0.f, 0.f, 0.f};

        // --- software-pipelined K loop: each B frag feeds 2 M-subtiles ---
        ABfrag b0, b1, a0m0, a0m1, a1m0, a1m1;
        loadB(b0, pW, nt_g, 0, lane);
        loadA(a0m0, arow0, 0);
        loadA(a0m1, arow1, 0);
        #pragma clang loop unroll(disable)
        for (int kt = 0; kt < KT_; kt += 2) {
            loadB(b1, pW, nt_g, kt + 1, lane);
            loadA(a1m0, arow0, kt + 1);
            loadA(a1m1, arow1, kt + 1);
            acc0 = __builtin_amdgcn_wmma_f32_16x16x32_bf16(
                false, a0m0.v, false, b0.v, (short)0, acc0, false, false);
            acc1 = __builtin_amdgcn_wmma_f32_16x16x32_bf16(
                false, a0m1.v, false, b0.v, (short)0, acc1, false, false);
            if (kt + 2 < KT_) {
                loadB(b0, pW, nt_g, kt + 2, lane);
                loadA(a0m0, arow0, kt + 2);
                loadA(a0m1, arow1, kt + 2);
            }
            acc0 = __builtin_amdgcn_wmma_f32_16x16x32_bf16(
                false, a1m0.v, false, b1.v, (short)0, acc0, false, false);
            acc1 = __builtin_amdgcn_wmma_f32_16x16x32_bf16(
                false, a1m1.v, false, b1.v, (short)0, acc1, false, false);
        }
        // extended tile (kt == 32): loc + dec_proj + b_enc
        {
            ABfrag bb;
            int koff = lh * 8;
            #pragma unroll
            for (int j = 0; j < 8; ++j)
                bb.s[j]     = f2bf(extB(koff + j, Watt, decp_b, benc, n));
            #pragma unroll
            for (int j = 0; j < 8; ++j)
                bb.s[j + 8] = f2bf(extB(16 + koff + j, Watt, decp_b, benc, n));
            loadA(a0m0, arow0, KT_);
            loadA(a0m1, arow1, KT_);
            acc0 = __builtin_amdgcn_wmma_f32_16x16x32_bf16(
                false, a0m0.v, false, bb.v, (short)0, acc0, false, false);
            acc1 = __builtin_amdgcn_wmma_f32_16x16x32_bf16(
                false, a0m1.v, false, bb.v, (short)0, acc1, false, false);
        }
        float gv = gvec[n];
        #pragma unroll
        for (int r = 0; r < 8; ++r) {
            epart0[r] += gv * tanhf(acc0[r]);
            epart1[r] += gv * tanhf(acc1[r]);
        }
    }
    // accumulator tile layout: lane -> N = l15, rows M = lh*8 + r (+16 for sub 1)
    #pragma unroll
    for (int r = 0; r < 8; ++r) {
        atomicAdd(&lsE[lh * 8 + r],      epart0[r]);
        atomicAdd(&lsE[16 + lh * 8 + r], epart1[r]);
    }
    __syncthreads();
    if (tid < MTILE) e_out[(size_t)b * T_ + t0 + tid] = lsE[tid];
}

// ---------------------------------------------------------------------------
// 5) masked softmax over T (scale 2.0); padded positions -> 0
__global__ __launch_bounds__(256) void k_softmax(
    const float* __restrict__ e, const int* __restrict__ lens,
    float* __restrict__ wout)
{
    __shared__ float red[256];
    int b = blockIdx.x, tid = threadIdx.x;
    int len = lens[b];
    const float* eb = e + (size_t)b * T_;

    float mx = -INFINITY;
    for (int t = tid; t < T_; t += 256)
        if (t < len) mx = fmaxf(mx, SCALING * eb[t]);
    red[tid] = mx; __syncthreads();
    for (int s = 128; s > 0; s >>= 1) {
        if (tid < s) red[tid] = fmaxf(red[tid], red[tid + s]);
        __syncthreads();
    }
    mx = red[0]; __syncthreads();

    float sm = 0.f;
    for (int t = tid; t < T_; t += 256)
        if (t < len) sm += __expf(SCALING * eb[t] - mx);
    red[tid] = sm; __syncthreads();
    for (int s = 128; s > 0; s >>= 1) {
        if (tid < s) red[tid] += red[tid + s];
        __syncthreads();
    }
    float inv = 1.f / red[0];

    for (int t = tid; t < T_; t += 256)
        wout[(size_t)b * T_ + t] =
            (t < len) ? __expf(SCALING * eb[t] - mx) * inv : 0.f;
}

// ---------------------------------------------------------------------------
// 6) c[b,d] = sum_t w[b,t] * enc[b,t,d]   (bandwidth-bound pass)
__global__ __launch_bounds__(256) void k_context(
    const float* __restrict__ enc, const float* __restrict__ wrow,
    float* __restrict__ cout)
{
    __shared__ float sw[T_];   // 8 KB
    int b = blockIdx.y;
    int d = blockIdx.x * 256 + threadIdx.x;
    for (int i = threadIdx.x; i < T_; i += 256)
        sw[i] = wrow[(size_t)b * T_ + i];
    __syncthreads();
    const float* eb = enc + (size_t)b * T_ * E_ + d;
    float s = 0.f;
    #pragma unroll 4
    for (int t = 0; t < T_; ++t) s += sw[t] * eb[(size_t)t * E_];
    cout[(size_t)b * E_ + d] = s;
}

// ---------------------------------------------------------------------------
extern "C" void kernel_launch(void* const* d_in, const int* in_sizes, int n_in,
                              void* d_out, int out_size, void* d_ws, size_t ws_size,
                              hipStream_t stream)
{
    const float* enc_hs  = (const float*)d_in[0];   // B,T,E
    const int*   lens    = (const int*)  d_in[1];   // B
    const float* dec_z   = (const float*)d_in[2];   // B,DUNITS
    const float* attprev = (const float*)d_in[3];   // B,T
    const float* W_enc   = (const float*)d_in[4];   // E,D
    const float* b_enc   = (const float*)d_in[5];   // D
    const float* W_dec   = (const float*)d_in[6];   // DUNITS,D
    const float* W_att   = (const float*)d_in[7];   // NCH,D
    const float* conv_w  = (const float*)d_in[8];   // NCH,1,FLT
    const float* gvec_w  = (const float*)d_in[9];   // D
    // gvec_b (d_in[10]) is an additive constant before softmax -> no effect

    // workspace layout
    char* ws = (char*)d_ws;
    float* decp = (float*)ws;                                     // 128 KB
    float* acv  = (float*)(ws + (size_t)(128 << 10));             // 4 MB
    float* e_ws = (float*)(ws + (size_t)(128 << 10) + (4 << 20)); // 256 KB
    unsigned char* pW = (unsigned char*)(ws + (size_t)(128 << 10) + (4 << 20)
                                            + (256 << 10));      // 2 MB

    float* c_out = (float*)d_out;          // B x E
    float* w_out = c_out + (size_t)B_ * E_;// B x T

    k_decproj<<<dim3(D_ / 256, B_), 256, 0, stream>>>(dec_z, W_dec, decp);
    k_conv   <<<dim3(T_ / 256, B_), 256, 0, stream>>>(attprev, conv_w, acv);
    k_packW  <<<dim3(256), 256, 0, stream>>>(W_enc, pW);
    k_escore <<<dim3(T_ / MTILE, B_), 256, 0, stream>>>(
        enc_hs, gvec_w, W_att, b_enc, decp, acv, pW, e_ws);
    k_softmax<<<dim3(B_), 256, 0, stream>>>(e_ws, lens, w_out);
    k_context<<<dim3(E_ / 256, B_), 256, 0, stream>>>(enc_hs, w_out, c_out);
}
